// CoWindowAttention_80384607912553
// MI455X (gfx1250) — compile-verified, matrix operand order checked by reading, not run on hardware
//
#include <hip/hip_runtime.h>
#include <cstddef>

// ---------------- problem constants ----------------
#define NB   256   // big/query tokens per window (WU*WU)
#define NS   64    // small/key tokens per window (W*W)
#define SFD  128   // sf_dim
#define BFD  256   // bf_dim
#define HEADS 4
#define HDIM 32    // head_dim
#define CHUNK 32   // query rows processed per chunk
#define NWIN 1024
static constexpr float SCALE_F = 0.17677669529663687f; // 32^-0.5

// ---------------- vector types ----------------
typedef __attribute__((ext_vector_type(16))) __bf16  v16bf;
typedef __attribute__((ext_vector_type(8)))  __bf16  bf16x8;
typedef __attribute__((ext_vector_type(8)))  float   v8f;

// hardware fp32 -> bf16 (v_cvt, packed where the compiler can)
__device__ __forceinline__ __bf16 tobf(float f) { return (__bf16)f; }

// A/B fragment (16x32 bf16, wave32 layout):
// lane<16 : row m=lane,    K = {k0+0..7,  k0+16..23}
// lane>=16: row m=lane-16, K = {k0+8..15, k0+24..31}
// source: __bf16 row-major, row stride ld (elements). Works for LDS and global.
__device__ __forceinline__ v16bf load_frag_bf(const __bf16* base, int row0,
                                              int ld, int k0, int lane) {
  int m = lane & 15, half = lane >> 4;
  const __bf16* p = base + (size_t)(row0 + m) * ld + k0 + half * 8;
  bf16x8 lo = *(const bf16x8*)p;
  bf16x8 hi = *(const bf16x8*)(p + 16);
  v16bf t;
#pragma unroll
  for (int i = 0; i < 8; ++i) { t[i] = lo[i]; t[i + 8] = hi[i]; }
  return t;
}

// same fragment but sourced from fp32 row-major, converting on the fly (hw cvt)
__device__ __forceinline__ v16bf load_frag_f32(const float* base, int row0,
                                               int ld, int k0, int lane) {
  int m = lane & 15, half = lane >> 4;
  const float* p = base + (size_t)(row0 + m) * ld + k0 + half * 8;
  v16bf t;
#pragma unroll
  for (int i = 0; i < 8; ++i) { t[i] = tobf(p[i]); t[i + 8] = tobf(p[16 + i]); }
  return t;
}

__device__ __forceinline__ v8f wmma_bf16(v16bf a, v16bf b, v8f c) {
  return __builtin_amdgcn_wmma_f32_16x16x32_bf16(false, a, false, b,
                                                 (short)0, c, false, false);
}

// ================= precompute kernels =================

// Wbqt[j][c] = bf16( sum_t W1[c][t] * Wqkv[t][j] )  (folded big->Q weight, transposed)
// bqf[j]    = b1 @ Wqkv[:, j] + bqkv[j]
__global__ void prep_fold(const float* __restrict__ W1, const float* __restrict__ b1,
                          const float* __restrict__ Wqkv, const float* __restrict__ bqkv,
                          __bf16* __restrict__ Wbqt, float* __restrict__ bqf) {
  int j = blockIdx.x;    // 0..127
  int c = threadIdx.x;   // 0..255
  float s = 0.f;
  for (int t = 0; t < SFD; ++t) s += W1[c * SFD + t] * Wqkv[t * (3 * SFD) + j];
  Wbqt[(size_t)j * BFD + c] = tobf(s);
  if (c == 0) {
    float sb = bqkv[j];
    for (int t = 0; t < SFD; ++t) sb += b1[t] * Wqkv[t * (3 * SFD) + j];
    bqf[j] = sb;
  }
}

// transposed bf16 copies of Wk, Wv, W2 (so B-fragments read contiguous runs)
__global__ void prep_wts(const float* __restrict__ Wqkv, const float* __restrict__ W2,
                         __bf16* __restrict__ Wkt, __bf16* __restrict__ Wvt,
                         __bf16* __restrict__ W2t) {
  int j = blockIdx.x;    // 0..255
  int k = threadIdx.x;   // 0..127
  if (j < SFD) {
    Wkt[(size_t)j * SFD + k] = tobf(Wqkv[k * (3 * SFD) + SFD + j]);
    Wvt[(size_t)j * SFD + k] = tobf(Wqkv[k * (3 * SFD) + 2 * SFD + j]);
  }
  W2t[(size_t)j * SFD + k] = tobf(W2[k * BFD + j]);
}

// dense relative-position bias: bias_full[h][q][k]
__global__ void prep_bias(const float* __restrict__ table, float* __restrict__ bias_full) {
  int q = blockIdx.x;             // 0..255
  int kk = threadIdx.x & 63;      // 0..63
  int h = threadIdx.x >> 6;       // 0..3
  int ch = q >> 4, cw = q & 15;   // WU=16 grid
  int sh = kk >> 3, sw = kk & 7;  // W=8 grid
  int idx = (ch - sh + 7) * 15 + (cw - sw + 7);
  bias_full[((size_t)h * NB + q) * NS + kk] = table[idx * HEADS + h];
}

// ================= fused attention kernel =================
__global__ __launch_bounds__(256)
void cowin_attn(const float* __restrict__ big_x, const float* __restrict__ small_x,
                const __bf16* __restrict__ Wbqt, const float* __restrict__ bqf,
                const __bf16* __restrict__ Wkt, const __bf16* __restrict__ Wvt,
                const __bf16* __restrict__ W2t, const float* __restrict__ bias_full,
                const float* __restrict__ bqkv, const float* __restrict__ b2,
                float* __restrict__ out) {
  __shared__ __bf16 sK[NS][SFD];              // 16 KB, K row-major (keys x d)
  __shared__ __bf16 sVt[SFD][NS];             // 16 KB, V transposed (d x keys)
  __shared__ __bf16 sQO[CHUNK][SFD];          //  8 KB, Q slab, reused as O slab
  __shared__ __bf16 sA[HEADS][CHUNK][NS];     // 16 KB, softmaxed attn (bf16)

  const int b    = blockIdx.x;
  const int tid  = threadIdx.x;
  const int lane = tid & 31;
  const int wave = tid >> 5;          // 0..7
  const int m16  = lane & 15;
  const int half = lane >> 4;

  const float* bigB   = big_x   + (size_t)b * NB * BFD;
  const float* smallB = small_x + (size_t)b * NS * SFD;
  float*       outB   = out     + (size_t)b * NB * BFD;

  // ---------- Phase 1: K = small@Wk + bk, V = small@Wv + bv ----------
  {
    const int j0 = wave * 16;  // this wave's 16 output columns
#pragma unroll
    for (int rg = 0; rg < 4; ++rg) {        // 4 rowgroups of 16 keys
      v8f accK, accV;
      const float bk = bqkv[SFD + j0 + m16];
      const float bv = bqkv[2 * SFD + j0 + m16];
#pragma unroll
      for (int r = 0; r < 8; ++r) { accK[r] = bk; accV[r] = bv; }
#pragma unroll
      for (int ks = 0; ks < 4; ++ks) {      // reduction over 128
        v16bf a  = load_frag_f32(smallB, rg * 16, SFD, ks * 32, lane);
        v16bf wk = load_frag_bf(Wkt, j0, SFD, ks * 32, lane);
        v16bf wv = load_frag_bf(Wvt, j0, SFD, ks * 32, lane);
        accK = wmma_bf16(a, wk, accK);
        accV = wmma_bf16(a, wv, accV);
      }
#pragma unroll
      for (int r = 0; r < 8; ++r) {         // C layout: row = r + 8*half, col = m16
        int row = rg * 16 + r + 8 * half;
        sK[row][j0 + m16]  = tobf(accK[r]);
        sVt[j0 + m16][row] = tobf(accV[r]); // store V transposed
      }
    }
  }
  __syncthreads();

  // ---------- per-chunk loop over 256 query rows ----------
  for (int ch = 0; ch < NB / CHUNK; ++ch) {
    const int row0 = ch * CHUNK;

    if (ch + 1 < NB / CHUNK) {              // prefetch next chunk of big_x
      __builtin_prefetch(bigB + (size_t)(row0 + CHUNK) * BFD + tid * 32, 0, 1);
    }

    // ---- Phase 2: Q slab = big_x[rows] @ Wbq + bqf, scaled, -> sQO ----
    {
      const int j0 = wave * 16;
#pragma unroll
      for (int rg = 0; rg < 2; ++rg) {
        v8f acc;
        const float bq = bqf[j0 + m16];
#pragma unroll
        for (int r = 0; r < 8; ++r) acc[r] = bq;
#pragma unroll
        for (int ks = 0; ks < 8; ++ks) {    // reduction over 256
          v16bf a = load_frag_f32(bigB, row0 + rg * 16, BFD, ks * 32, lane);
          v16bf w = load_frag_bf(Wbqt, j0, BFD, ks * 32, lane);
          acc = wmma_bf16(a, w, acc);
        }
#pragma unroll
        for (int r = 0; r < 8; ++r)
          sQO[rg * 16 + r + 8 * half][j0 + m16] = tobf(acc[r] * SCALE_F);
      }
    }
    __syncthreads();

    // ---- Phase 3: attn = QK^T + bias, softmax (registers), -> sA ----
    {
      const int rg = wave & 1;              // 8 waves = 2 rowgroups x 4 heads
      const int h  = wave >> 1;
      v16bf a = load_frag_bf(&sQO[0][0], rg * 16, SFD, h * HDIM, lane);
      v8f t[4];
#pragma unroll
      for (int ct = 0; ct < 4; ++ct) {      // 4 key tiles of 16
        v8f c;
        const float* bp = bias_full + ((size_t)h * NB + row0 + rg * 16) * NS + ct * 16 + m16;
#pragma unroll
        for (int r = 0; r < 8; ++r) c[r] = bp[(size_t)(r + 8 * half) * NS];
        v16bf kb = load_frag_bf(&sK[0][0], ct * 16, SFD, h * HDIM, lane);
        t[ct] = wmma_bf16(a, kb, c);        // HD=32 -> single WMMA
      }
      // per-row softmax: row lives in one VGPR index across 16 lanes x 4 tiles
#pragma unroll
      for (int r = 0; r < 8; ++r) {
        float mx = fmaxf(fmaxf(t[0][r], t[1][r]), fmaxf(t[2][r], t[3][r]));
        mx = fmaxf(mx, __shfl_xor(mx, 1, 16));
        mx = fmaxf(mx, __shfl_xor(mx, 2, 16));
        mx = fmaxf(mx, __shfl_xor(mx, 4, 16));
        mx = fmaxf(mx, __shfl_xor(mx, 8, 16));
        float s = 0.f;
#pragma unroll
        for (int ct = 0; ct < 4; ++ct) { t[ct][r] = __expf(t[ct][r] - mx); s += t[ct][r]; }
        s += __shfl_xor(s, 1, 16);
        s += __shfl_xor(s, 2, 16);
        s += __shfl_xor(s, 4, 16);
        s += __shfl_xor(s, 8, 16);
        const float inv = 1.0f / s;
        const int row = rg * 16 + r + 8 * half;
#pragma unroll
        for (int ct = 0; ct < 4; ++ct)
          sA[h][row][ct * 16 + m16] = tobf(t[ct][r] * inv);
      }
    }
    __syncthreads();

    // ---- Phase 4: O = attn @ V  -> sQO (aliased; Q is dead now) ----
    {
      const int rg = wave & 1;
      const int h  = wave >> 1;
#pragma unroll
      for (int ct = 0; ct < 2; ++ct) {      // 2 d-tiles of 16
        v8f acc = {};
#pragma unroll
        for (int ks = 0; ks < 2; ++ks) {    // reduction over 64 keys
          v16bf a = load_frag_bf(&sA[h][0][0], rg * 16, NS, ks * 32, lane);
          v16bf v = load_frag_bf(&sVt[0][0], h * HDIM + ct * 16, NS, ks * 32, lane);
          acc = wmma_bf16(a, v, acc);
        }
#pragma unroll
        for (int r = 0; r < 8; ++r)
          sQO[rg * 16 + r + 8 * half][h * HDIM + ct * 16 + m16] = tobf(acc[r]);
      }
    }
    __syncthreads();

    // ---- Phase 5: Y = O @ W2 + b2 -> HBM ----
    {
      const int c0 = wave * 32;             // this wave's 32 output columns
#pragma unroll
      for (int rg = 0; rg < 2; ++rg) {
#pragma unroll
        for (int ct = 0; ct < 2; ++ct) {
          v8f acc;
          const float bb = b2[c0 + ct * 16 + m16];
#pragma unroll
          for (int r = 0; r < 8; ++r) acc[r] = bb;
#pragma unroll
          for (int ks = 0; ks < 4; ++ks) {  // reduction over 128
            v16bf a = load_frag_bf(&sQO[0][0], rg * 16, SFD, ks * 32, lane);
            v16bf w = load_frag_bf(W2t, c0 + ct * 16, SFD, ks * 32, lane);
            acc = wmma_bf16(a, w, acc);
          }
#pragma unroll
          for (int r = 0; r < 8; ++r)
            outB[(size_t)(row0 + rg * 16 + r + 8 * half) * BFD + c0 + ct * 16 + m16] = acc[r];
        }
      }
    }
    __syncthreads();
  }
}

// ================= launcher =================
extern "C" void kernel_launch(void* const* d_in, const int* in_sizes, int n_in,
                              void* d_out, int out_size, void* d_ws, size_t ws_size,
                              hipStream_t stream) {
  (void)in_sizes; (void)n_in; (void)out_size; (void)ws_size;
  const float* big_x   = (const float*)d_in[0];
  const float* small_x = (const float*)d_in[1];
  const float* W1      = (const float*)d_in[2];
  const float* b1      = (const float*)d_in[3];
  const float* Wqkv    = (const float*)d_in[4];
  const float* bqkv    = (const float*)d_in[5];
  const float* W2      = (const float*)d_in[6];
  const float* b2      = (const float*)d_in[7];
  const float* table   = (const float*)d_in[8];
  float* out = (float*)d_out;

  char* ws = (char*)d_ws;
  __bf16* Wbqt = (__bf16*)ws; ws += (size_t)SFD * BFD * 2;  // 64 KB
  __bf16* Wkt  = (__bf16*)ws; ws += (size_t)SFD * SFD * 2;  // 32 KB
  __bf16* Wvt  = (__bf16*)ws; ws += (size_t)SFD * SFD * 2;  // 32 KB
  __bf16* W2t  = (__bf16*)ws; ws += (size_t)BFD * SFD * 2;  // 64 KB
  float* bqf       = (float*)ws; ws += (size_t)SFD * 4;     // 512 B
  float* bias_full = (float*)ws;                            // 256 KB

  prep_fold<<<SFD, BFD, 0, stream>>>(W1, b1, Wqkv, bqkv, Wbqt, bqf);
  prep_wts<<<BFD, SFD, 0, stream>>>(Wqkv, W2, Wkt, Wvt, W2t);
  prep_bias<<<NB, HEADS * NS, 0, stream>>>(table, bias_full);
  cowin_attn<<<NWIN, 256, 0, stream>>>(big_x, small_x, Wbqt, bqf, Wkt, Wvt, W2t,
                                       bias_full, bqkv, b2, out);
}
